// RPN_15118284882404
// MI455X (gfx1250) — matrix operand order repaired and checked by reading, not compiled, not run on hardware
//
#include <hip/hip_runtime.h>

typedef unsigned short u16;
typedef unsigned int   u32;
typedef __attribute__((ext_vector_type(16))) _Float16 v16h;
typedef __attribute__((ext_vector_type(8)))  float    v8f;

union Frag { v16h v; uint4 u[2]; };

#define FH 64
#define FW 64
#define CIN 1024
#define CMID 256
#define NPIX (FH*FW)          // 4096
#define KTILES 288            // 9216 / 32
#define KSPLIT 3              // one 3x3 filter row per split -> no div/mod in hot loop
#define KANCH 9
#define NA (NPIX*KANCH)       // 36864
#define PRE_NMS 6000
#define POST_NMS 2000
#define MASKW 188             // ceil(6000/32)
#define NGT 20
#define FGN 32
#define BGN 96
#define NROIS 128
#define BBOX_CLIP 4.135f

// ---------------- workspace layout (all 16B aligned) ----------------
constexpr size_t OFF_XPAD = 0;                                  // 66*66*1024 f16
constexpr size_t SZ_XPAD  = (size_t)66*66*1024*2;               // 8,921,088
constexpr size_t OFF_BPK  = OFF_XPAD + SZ_XPAD;                 // weights, WMMA B layout
constexpr size_t SZ_BPK   = (size_t)KTILES*16*32*16*2;          // 4,718,592
constexpr size_t OFF_PART = OFF_BPK + SZ_BPK;                   // split-K partials f32
constexpr size_t SZ_PART  = (size_t)KSPLIT*NPIX*CMID*4;         // 12,582,912
constexpr size_t OFF_S    = OFF_PART + SZ_PART;                 // conv out f32
constexpr size_t SZ_S     = (size_t)NPIX*CMID*4;                // 4,194,304
constexpr size_t OFF_SC   = OFF_S + SZ_S;                       // fg scores
constexpr size_t SZ_SC    = (size_t)NA*4;
constexpr size_t OFF_PR   = OFF_SC + SZ_SC;                     // proposals (float4)
constexpr size_t SZ_PR    = (size_t)NA*16;
constexpr size_t OFF_RK   = OFF_PR + SZ_PR;                     // ranks
constexpr size_t SZ_RK    = (size_t)NA*4;
constexpr size_t OFF_TB   = OFF_RK + SZ_RK;                     // top-6000 boxes (float4)
constexpr size_t SZ_TB    = (size_t)PRE_NMS*16;
constexpr size_t OFF_MK   = OFF_TB + SZ_TB;                     // NMS bitmask
constexpr size_t SZ_MK    = (size_t)PRE_NMS*MASKW*4;            // 4,512,000
constexpr size_t OFF_SEL  = OFF_MK + SZ_MK;                     // sel[2000]
constexpr size_t SZ_SEL   = (size_t)POST_NMS*4;
constexpr size_t OFF_RB   = OFF_SEL + SZ_SEL;                   // roi boxes (float4)
constexpr size_t SZ_RB    = (size_t)POST_NMS*16;
constexpr size_t OFF_MO   = OFF_RB + SZ_RB;                     // max overlap
constexpr size_t SZ_MO    = (size_t)POST_NMS*4;
constexpr size_t OFF_FG   = OFF_MO + SZ_MO;                     // fg sel (32)
constexpr size_t OFF_BG   = OFF_FG + 256;                       // bg sel (96)

__device__ inline u16 f2h(float f) {
    _Float16 h = (_Float16)f;
    return __builtin_bit_cast(u16, h);
}

// ---------------- stage 0a: pad + convert input to f16 ----------------
__global__ __launch_bounds__(256) void prep_xpad(const float* __restrict__ X, u16* __restrict__ xpad) {
    int idx = blockIdx.x * 256 + threadIdx.x;       // 66*66*1024 total
    int c = idx & (CIN - 1);
    int r = idx >> 10;
    int x = r % 66, y = r / 66;
    u16 v = 0;
    if (y >= 1 && y <= FH && x >= 1 && x <= FW)
        v = f2h(X[((size_t)(y - 1) * FW + (x - 1)) * CIN + c]);
    xpad[idx] = v;
}

// ---------------- stage 0b: swizzle weights into WMMA-B fragment layout -------
// B-frag (16-bit, 32x16): lane n in [0,16): col n, vgpr j<4 holds K=2j,2j+1, j>=4 K=8+2j,9+2j ; lanes 16..31: K+8
// Stored per (ktile, ntile): 32 lanes x 8 dwords, lane-major.
__global__ __launch_bounds__(256) void prep_bpk(const float* __restrict__ W, u32* __restrict__ bpk) {
    int idx = blockIdx.x * 256 + threadIdx.x;       // KTILES*16*32*8 dwords
    int j    = idx & 7;
    int lane = (idx >> 3) & 31;
    int t    = idx >> 8;
    int nt   = t & 15;
    int kt   = t >> 4;
    int n  = nt * 16 + (lane & 15);
    int k0 = kt * 32 + ((lane >> 4) << 3) + 2 * j + ((j >> 2) << 3);
    u16 lo = f2h(W[(size_t)k0 * CMID + n]);
    u16 hi = f2h(W[(size_t)(k0 + 1) * CMID + n]);
    bpk[idx] = (u32)lo | ((u32)hi << 16);
}

// ---------------- stage 1: 3x3 conv as implicit GEMM via WMMA ----------------
// M=4096, N=256, K=9216. Block tile 64Mx256N, 8 waves, wave tile 32x64.
// Split-K=3 (one filter row per split) into fp32 partials (deterministic).
struct KTile { Frag a0, a1, b0, b1, b2, b3; };

// ci = K-slice index within this (dy,dx) tap, 0..31. Addresses are pure
// base + ci*const so the hot loop is free of mul/div address math.
__device__ __forceinline__ void load_ktile(KTile& t, int ci,
                                           const u16* __restrict__ a0r,
                                           const u16* __restrict__ a1r,
                                           const u16* __restrict__ bb0) {
    const u16* a0 = a0r + ci * 32;
    const u16* a1 = a1r + ci * 32;
    t.a0.u[0] = *(const uint4*)(a0);
    t.a0.u[1] = *(const uint4*)(a0 + 16);
    t.a1.u[0] = *(const uint4*)(a1);
    t.a1.u[1] = *(const uint4*)(a1 + 16);
    const u16* bb = bb0 + (size_t)ci * 8192;        // 16 tiles * 512 elems per k-tile
    t.b0.u[0] = *(const uint4*)(bb);
    t.b0.u[1] = *(const uint4*)(bb + 8);
    t.b1.u[0] = *(const uint4*)(bb + 512);
    t.b1.u[1] = *(const uint4*)(bb + 520);
    t.b2.u[0] = *(const uint4*)(bb + 1024);
    t.b2.u[1] = *(const uint4*)(bb + 1032);
    t.b3.u[0] = *(const uint4*)(bb + 1536);
    t.b3.u[1] = *(const uint4*)(bb + 1544);
}

#define WMMA8(T)                                                                              \
    c0 = __builtin_amdgcn_wmma_f32_16x16x32_f16(false, (T).a0.v, false, (T).b0.v, (short)0, c0, false, false); \
    c1 = __builtin_amdgcn_wmma_f32_16x16x32_f16(false, (T).a0.v, false, (T).b1.v, (short)0, c1, false, false); \
    c2 = __builtin_amdgcn_wmma_f32_16x16x32_f16(false, (T).a0.v, false, (T).b2.v, (short)0, c2, false, false); \
    c3 = __builtin_amdgcn_wmma_f32_16x16x32_f16(false, (T).a0.v, false, (T).b3.v, (short)0, c3, false, false); \
    c4 = __builtin_amdgcn_wmma_f32_16x16x32_f16(false, (T).a1.v, false, (T).b0.v, (short)0, c4, false, false); \
    c5 = __builtin_amdgcn_wmma_f32_16x16x32_f16(false, (T).a1.v, false, (T).b1.v, (short)0, c5, false, false); \
    c6 = __builtin_amdgcn_wmma_f32_16x16x32_f16(false, (T).a1.v, false, (T).b2.v, (short)0, c6, false, false); \
    c7 = __builtin_amdgcn_wmma_f32_16x16x32_f16(false, (T).a1.v, false, (T).b3.v, (short)0, c7, false, false)

__global__ __launch_bounds__(256) void conv3x3_wmma(const u16* __restrict__ xpad,
                                                    const u16* __restrict__ bpk,
                                                    float* __restrict__ part) {
    int m0   = blockIdx.x * 64;
    int dy   = blockIdx.y;           // split-K: one filter row per y-block
    int wave = threadIdx.x >> 5;
    int lane = threadIdx.x & 31;
    int mw   = (wave >> 2) * 32;     // 0 or 32
    int nw   = (wave & 3) * 64;      // 0,64,128,192
    int nt0  = nw >> 4;
    v8f c0 = {}, c1 = {}, c2 = {}, c3 = {}, c4 = {}, c5 = {}, c6 = {}, c7 = {};

    int pix0 = m0 + mw + (lane & 15);
    int pix1 = pix0 + 16;
    int py0 = pix0 >> 6, px0 = pix0 & 63;
    int py1 = pix1 >> 6, px1 = pix1 & 63;
    int hiHalf = (lane >> 4) << 3;   // +8 K-elements for lanes 16..31 (A layout)

    // per-tap bases: A row for (dy,dx) is xpad[(py+dy)*66 + (px+dx)]
    const u16* a0base = xpad + ((size_t)((py0 + dy) * 66 + px0)) * CIN + hiHalf;
    const u16* a1base = xpad + ((size_t)((py1 + dy) * 66 + px1)) * CIN + hiHalf;
    // global k-tile base for this split: kt = (dy*3+dx)*32 + ci
    const u16* bblane = bpk + ((size_t)nt0 * 32 + lane) * 16;

    for (int dx = 0; dx < 3; ++dx) {
        const u16* a0r = a0base + (size_t)dx * CIN;
        const u16* a1r = a1base + (size_t)dx * CIN;
        const u16* bb0 = bblane + (size_t)(dy * 3 + dx) * 32 * 8192;
        KTile cur, nxt;
        load_ktile(cur, 0, a0r, a1r, bb0);
        for (int ci = 0; ci < 32; ci += 2) {
            load_ktile(nxt, ci + 1, a0r, a1r, bb0);
            WMMA8(cur);
            int cin2 = (ci + 2 < 32) ? ci + 2 : 31;   // harmless dummy reload on last pair
            load_ktile(cur, cin2, a0r, a1r, bb0);
            WMMA8(nxt);
        }
    }

    // D layout: vgpr r -> M = r (lanes<16) / 8+r (lanes>=16), N = lane&15
    float* Sp = part + (size_t)dy * NPIX * CMID;
    int crow = (lane < 16) ? 0 : 8;
    int ccol = lane & 15;
#pragma unroll
    for (int j = 0; j < 4; ++j) {
        v8f accA = (j == 0) ? c0 : (j == 1) ? c1 : (j == 2) ? c2 : c3;
        v8f accB = (j == 0) ? c4 : (j == 1) ? c5 : (j == 2) ? c6 : c7;
#pragma unroll
        for (int r = 0; r < 8; ++r) {
            Sp[(size_t)(m0 + mw + crow + r) * CMID + nw + j * 16 + ccol]      = accA[r];
            Sp[(size_t)(m0 + mw + 16 + crow + r) * CMID + nw + j * 16 + ccol] = accB[r];
        }
    }
}

__global__ __launch_bounds__(256) void reduce_parts(const float* __restrict__ part, float* __restrict__ S) {
    int i = blockIdx.x * 256 + threadIdx.x;          // NPIX*CMID total
    const size_t STRIDE = (size_t)NPIX * CMID;
    S[i] = (part[i] + part[i + STRIDE]) + part[i + 2 * STRIDE];
}

// ---------------- stage 2: 1x1 heads + softmax + anchor decode + clip --------
__global__ __launch_bounds__(64) void rpn_head(const float* __restrict__ S,
    const float* __restrict__ bsh, const float* __restrict__ wcls, const float* __restrict__ bcls,
    const float* __restrict__ wbb, const float* __restrict__ bbb,
    float* __restrict__ scores, float4* __restrict__ props) {
    __shared__ float srow[CMID];
    __shared__ float outs[54];
    int pix = blockIdx.x, t = threadIdx.x;
    for (int i = t; i < CMID; i += 64) srow[i] = S[(size_t)pix * CMID + i] + bsh[i];
    __syncthreads();
    if (t < 54) {
        float acc;
        if (t < 18) { acc = bcls[t]; for (int cc = 0; cc < CMID; ++cc) acc += srow[cc] * wcls[cc * 18 + t]; }
        else { int u = t - 18; acc = bbb[u]; for (int cc = 0; cc < CMID; ++cc) acc += srow[cc] * wbb[cc * 36 + u]; }
        outs[t] = acc;
    }
    __syncthreads();
    if (t < KANCH) {
        int k = t;
        float s0 = outs[2 * k], s1 = outs[2 * k + 1];
        float fg = 1.0f / (1.0f + expf(s0 - s1));   // softmax over 2
        float dxv = outs[18 + 4 * k + 0];
        float dyv = outs[18 + 4 * k + 1];
        float dwv = fminf(fmaxf(outs[18 + 4 * k + 2], -BBOX_CLIP), BBOX_CLIP);
        float dhv = fminf(fmaxf(outs[18 + 4 * k + 3], -BBOX_CLIP), BBOX_CLIP);
        int py = pix >> 6, px = pix & 63;
        int rr = k / 3, sc = k - 3 * rr;
        float ratio = (rr == 0) ? 0.5f : (rr == 1) ? 1.0f : 2.0f;
        float scale = (sc == 0) ? 8.0f : (sc == 1) ? 16.0f : 32.0f;
        float ws0 = rintf(sqrtf(256.0f / ratio));   // RNE matches np.round
        float hs0 = rintf(ws0 * ratio);
        float aw = ws0 * scale, ah = hs0 * scale;
        float ctr = 7.5f;
        float x1 = ctr - 0.5f * (aw - 1.0f) + px * 16.0f;
        float y1 = ctr - 0.5f * (ah - 1.0f) + py * 16.0f;
        float x2 = ctr + 0.5f * (aw - 1.0f) + px * 16.0f;
        float y2 = ctr + 0.5f * (ah - 1.0f) + py * 16.0f;
        float w = x2 - x1 + 1.0f, h = y2 - y1 + 1.0f;
        float cx = x1 + 0.5f * w, cy = y1 + 0.5f * h;
        float pcx = dxv * w + cx, pcy = dyv * h + cy;
        float pw = expf(dwv) * w, ph = expf(dhv) * h;
        float X1 = fminf(fmaxf(pcx - 0.5f * pw, 0.0f), 1023.0f);
        float Y1 = fminf(fmaxf(pcy - 0.5f * ph, 0.0f), 1023.0f);
        float X2 = fminf(fmaxf(pcx + 0.5f * pw, 0.0f), 1023.0f);
        float Y2 = fminf(fmaxf(pcy + 0.5f * ph, 0.0f), 1023.0f);
        int gi = pix * KANCH + k;
        scores[gi] = fg;
        props[gi] = make_float4(X1, Y1, X2, Y2);
    }
}

// ---------------- stage 3: exact stable top-6000 via O(N^2) rank -------------
__global__ __launch_bounds__(256) void zero_rank(int* __restrict__ rk) {
    rk[blockIdx.x * 256 + threadIdx.x] = 0;
}

__global__ __launch_bounds__(256) void rank_scores(const float* __restrict__ sc, int* __restrict__ rk) {
    __shared__ float tile[256];
    int i = blockIdx.x * 256 + threadIdx.x;
    float si = sc[i];
    int j0 = blockIdx.y * 4096;
    int cnt = 0;
    for (int jb = 0; jb < 4096; jb += 256) {
        tile[threadIdx.x] = sc[j0 + jb + threadIdx.x];
        __syncthreads();
        for (int u = 0; u < 256; ++u) {
            float sj = tile[u]; int j = j0 + jb + u;
            cnt += ((sj > si) || (sj == si && j < i)) ? 1 : 0;
        }
        __syncthreads();
    }
    atomicAdd(&rk[i], cnt);
}

__global__ __launch_bounds__(256) void scatter_top(const int* __restrict__ rk,
                                                   const float4* __restrict__ props,
                                                   float4* __restrict__ tb) {
    int i = blockIdx.x * 256 + threadIdx.x;
    int r = rk[i];
    if (r < PRE_NMS) tb[r] = props[i];
}

// ---------------- stage 4: NMS suppression bitmask ---------------------------
__global__ __launch_bounds__(128) void nms_mask(const float4* __restrict__ tb, u32* __restrict__ mask) {
    __shared__ float4 cb[32];
    int i = blockIdx.x * 128 + threadIdx.x;
    int jb = blockIdx.y * 32;
    if (threadIdx.x < 32) {
        int j = jb + threadIdx.x;
        cb[threadIdx.x] = (j < PRE_NMS) ? tb[j] : make_float4(-1e8f, -1e8f, -1e8f, -1e8f);
    }
    __syncthreads();
    if (i >= PRE_NMS) return;
    float4 a = tb[i];
    float aarea = (a.z - a.x + 1.0f) * (a.w - a.y + 1.0f);
    u32 bits = 0;
    for (int u = 0; u < 32; ++u) {
        int j = jb + u;
        if (j <= i || j >= PRE_NMS) continue;
        float4 b = cb[u];
        float barea = (b.z - b.x + 1.0f) * (b.w - b.y + 1.0f);
        float iw = fmaxf(fminf(a.z, b.z) - fmaxf(a.x, b.x) + 1.0f, 0.0f);
        float ih = fmaxf(fminf(a.w, b.w) - fmaxf(a.y, b.y) + 1.0f, 0.0f);
        float inter = iw * ih;
        float iou = inter / (aarea + barea - inter);
        if (iou > 0.7f) bits |= (1u << u);
    }
    mask[(size_t)i * MASKW + blockIdx.y] = bits;
}

__device__ inline u32 pick6(u32 r0, u32 r1, u32 r2, u32 r3, u32 r4, u32 r5, int c) {
    switch (c) { case 0: return r0; case 1: return r1; case 2: return r2;
                 case 3: return r3; case 4: return r4; default: return r5; }
}

// single-wave greedy scan; removed-bitmap lives in 6 regs/lane (188 words)
__global__ __launch_bounds__(32) void nms_scan(const u32* __restrict__ mask, int* __restrict__ sel) {
    int lane = threadIdx.x;
    u32 r0 = 0, r1 = 0, r2 = 0, r3 = 0, r4 = 0, r5 = 0;
    int kcount = 0;
    for (int i = 0; i < PRE_NMS; ++i) {
        int w = i >> 5, c = w >> 5, src = w & 31;
        u32 word = pick6(r0, r1, r2, r3, r4, r5, c);
        u32 bc = (u32)__shfl((int)word, src, 32);
        bool removed = (bc >> (i & 31)) & 1u;
        if (!removed) {
            if (lane == 0 && kcount < POST_NMS) sel[kcount] = i;
            const u32* row = mask + (size_t)i * MASKW;
            r0 |= row[lane];
            r1 |= row[32 + lane];
            r2 |= row[64 + lane];
            r3 |= row[96 + lane];
            r4 |= row[128 + lane];
            if (lane < 28) r5 |= row[160 + lane];
            kcount++;
        }
        if (lane < 6 && i + 1 < PRE_NMS)
            __builtin_prefetch(mask + (size_t)(i + 1) * MASKW + lane * 32, 0, 1);
    }
    // fill remainder with removed indices, ascending (matches rank+argsort)
    int slot = (kcount < POST_NMS) ? kcount : POST_NMS;
    for (int w = 0; w < MASKW && slot < POST_NMS; ++w) {
        int c = w >> 5, src = w & 31;
        u32 word = pick6(r0, r1, r2, r3, r4, r5, c);
        u32 bc = (u32)__shfl((int)word, src, 32);
        int top = (w == MASKW - 1) ? 16 : 32;   // 187*32 + 16 = 6000
        for (int b = 0; b < top && slot < POST_NMS; ++b) {
            if ((bc >> b) & 1u) {
                if (lane == 0) sel[slot] = w * 32 + b;
                slot++;
            }
        }
    }
}

// ---------------- stage 5: gather rois + IoU vs GT ---------------------------
__global__ __launch_bounds__(128) void roi_iou(const int* __restrict__ sel, const float4* __restrict__ tb,
                                               const float* __restrict__ gt,
                                               float4* __restrict__ rb, float* __restrict__ mo) {
    int t = blockIdx.x * 128 + threadIdx.x;
    if (t >= POST_NMS) return;
    float4 a = tb[sel[t]];
    rb[t] = a;
    float aarea = (a.z - a.x + 1.0f) * (a.w - a.y + 1.0f);
    float best = 0.0f;
    for (int g = 0; g < NGT; ++g) {
        float bx1 = gt[g * 5 + 0], by1 = gt[g * 5 + 1], bx2 = gt[g * 5 + 2], by2 = gt[g * 5 + 3];
        float barea = (bx2 - bx1 + 1.0f) * (by2 - by1 + 1.0f);
        float iw = fmaxf(fminf(a.z, bx2) - fmaxf(a.x, bx1) + 1.0f, 0.0f);
        float ih = fmaxf(fminf(a.w, by2) - fmaxf(a.y, by1) + 1.0f, 0.0f);
        float inter = iw * ih;
        best = fmaxf(best, inter / (aarea + barea - inter));
    }
    mo[t] = best;
}

// ---------------- stage 6: fg/bg top-k via stable rank ----------------------
__global__ __launch_bounds__(256) void fgbg_select(const float* __restrict__ mo,
                                                   int* __restrict__ fgsel, int* __restrict__ bgsel) {
    __shared__ float tile[256];
    int t = blockIdx.x * 256 + threadIdx.x;
    float m = (t < POST_NMS) ? mo[t] : 0.0f;
    float fi = (m >= 0.5f) ? m : -1.0f;
    float bi = (m < 0.5f && m >= 0.1f) ? m : -1.0f;
    int rf = 0, rb_ = 0;
    for (int jb = 0; jb < POST_NMS; jb += 256) {
        int j = jb + threadIdx.x;
        tile[threadIdx.x] = (j < POST_NMS) ? mo[j] : -2.0f;
        __syncthreads();
        int lim = POST_NMS - jb; if (lim > 256) lim = 256;
        for (int u = 0; u < lim; ++u) {
            float mj = tile[u]; int j2 = jb + u;
            float fj = (mj >= 0.5f) ? mj : -1.0f;
            float bj = (mj < 0.5f && mj >= 0.1f) ? mj : -1.0f;
            rf  += ((fj > fi) || (fj == fi && j2 < t)) ? 1 : 0;
            rb_ += ((bj > bi) || (bj == bi && j2 < t)) ? 1 : 0;
        }
        __syncthreads();
    }
    if (t < POST_NMS) {
        if (rf  < FGN) fgsel[rf]  = t;
        if (rb_ < BGN) bgsel[rb_] = t;
    }
}

__global__ __launch_bounds__(128) void final_gather(const int* __restrict__ fgsel,
                                                    const int* __restrict__ bgsel,
                                                    const float4* __restrict__ rb,
                                                    float* __restrict__ out) {
    int t = threadIdx.x;
    if (t >= NROIS) return;
    int src = (t < FGN) ? fgsel[t] : bgsel[t - FGN];
    float4 b = rb[src];
    out[t * 5 + 0] = 0.0f;
    out[t * 5 + 1] = b.x;
    out[t * 5 + 2] = b.y;
    out[t * 5 + 3] = b.z;
    out[t * 5 + 4] = b.w;
}

extern "C" void kernel_launch(void* const* d_in, const int* in_sizes, int n_in,
                              void* d_out, int out_size, void* d_ws, size_t ws_size,
                              hipStream_t stream) {
    const float* X   = (const float*)d_in[0];
    const float* Wsh = (const float*)d_in[1];
    const float* bsh = (const float*)d_in[2];
    const float* wcl = (const float*)d_in[3];
    const float* bcl = (const float*)d_in[4];
    const float* wbb = (const float*)d_in[5];
    const float* bbb = (const float*)d_in[6];
    const float* gt  = (const float*)d_in[7];
    (void)in_sizes; (void)n_in; (void)out_size; (void)ws_size;

    char* ws = (char*)d_ws;
    u16*    xpad = (u16*)(ws + OFF_XPAD);
    u16*    bpk  = (u16*)(ws + OFF_BPK);
    float*  part = (float*)(ws + OFF_PART);
    float*  S    = (float*)(ws + OFF_S);
    float*  sc   = (float*)(ws + OFF_SC);
    float4* pr   = (float4*)(ws + OFF_PR);
    int*    rk   = (int*)(ws + OFF_RK);
    float4* tb   = (float4*)(ws + OFF_TB);
    u32*    mk   = (u32*)(ws + OFF_MK);
    int*    sel  = (int*)(ws + OFF_SEL);
    float4* rb   = (float4*)(ws + OFF_RB);
    float*  mo   = (float*)(ws + OFF_MO);
    int*    fg   = (int*)(ws + OFF_FG);
    int*    bg   = (int*)(ws + OFF_BG);

    prep_xpad   <<<(66 * 66 * 1024) / 256, 256, 0, stream>>>(X, xpad);
    prep_bpk    <<<(KTILES * 16 * 32 * 8) / 256, 256, 0, stream>>>(Wsh, (u32*)bpk);
    conv3x3_wmma<<<dim3(NPIX / 64, KSPLIT), 256, 0, stream>>>(xpad, bpk, part);
    reduce_parts<<<(NPIX * CMID) / 256, 256, 0, stream>>>(part, S);
    rpn_head    <<<NPIX, 64, 0, stream>>>(S, bsh, wcl, bcl, wbb, bbb, sc, pr);
    zero_rank   <<<NA / 256, 256, 0, stream>>>(rk);
    rank_scores <<<dim3(NA / 256, 9), 256, 0, stream>>>(sc, rk);
    scatter_top <<<NA / 256, 256, 0, stream>>>(rk, pr, tb);
    nms_mask    <<<dim3((PRE_NMS + 127) / 128, MASKW), 128, 0, stream>>>(tb, mk);
    nms_scan    <<<1, 32, 0, stream>>>(mk, sel);
    roi_iou     <<<(POST_NMS + 127) / 128, 128, 0, stream>>>(sel, tb, gt, rb, mo);
    fgbg_select <<<(POST_NMS + 255) / 256, 256, 0, stream>>>(mo, fg, bg);
    final_gather<<<1, 128, 0, stream>>>(fg, bg, rb, (float*)d_out);
}
